// TransformerDTI_841813590195
// MI455X (gfx1250) — matrix operand order repaired and checked by reading, not compiled, not run on hardware
//
#include <hip/hip_runtime.h>
#include <math.h>

// ---------------------------------------------------------------------------
// TransformerDTI backdoor-adjust forward for CDNA5 (gfx1250), v3.
//
// Strategy: convert/swizzle every GEMM operand ONCE into bf16 "fragment
// layout" (the exact per-lane VGPR layout of v_wmma_f32_16x16x32_bf16,
// CDNA5 ISA §7.12.2). Fragments are 32 lanes x 8 dwords,
// contiguous in memory, so the GEMM inner loop is just coalesced
// global_load_b128 -> v_wmma with no LDS, no barriers, no conversion VALU.
//
// A-frag (16x32, MxK): lane(0-15)=row r, dwords: K0-7,K16-23 ; lanes 16-31: K8-15,K24-31
// B-frag (32x16, KxN): lane(0-15)=col c, K0-7,K16-23 ; lanes 16-31: K8-15,K24-31
// C/D    (16x16 f32) : lane(0-15)=N, vgpr r -> M=r ; lanes 16-31 -> M=r+8
//
// Algebra: [x | att@K] @ Wpr == x@Wpr_top + att@(K@Wpr_bot), so Qa is never
// materialized. Q and att live only as bf16 fragment-packed tensors.
// ---------------------------------------------------------------------------

typedef __attribute__((ext_vector_type(16))) __bf16 v16bf;
typedef __attribute__((ext_vector_type(8)))  float  v8f;

#define NEG_INF_F (-1e10f)

// f32 pair -> packed bf16 (RNE), lo = first element
__device__ inline unsigned pack_bf16x2(float a, float b) {
  unsigned ua = __builtin_bit_cast(unsigned, a);
  unsigned ub = __builtin_bit_cast(unsigned, b);
  ua += 0x7FFFu + ((ua >> 16) & 1u);
  ub += 0x7FFFu + ((ub >> 16) & 1u);
  return (ua >> 16) | (ub & 0xFFFF0000u);
}

// fragment-slot helpers (dword granularity; one fragment = 256 dwords)
__device__ inline int a_slot(int r /*row%16*/, int kp /*(k%32)/2*/) {
  int lane = r + (((kp >> 2) & 1) << 4);
  int v    = (kp & 3) + ((kp >> 3) << 2);
  return lane * 8 + v;
}
__device__ inline int b_slot(int c /*col%16*/, int kp) {
  int lane = c + (((kp >> 2) & 1) << 4);
  int v    = (kp & 3) + ((kp >> 3) << 2);
  return lane * 8 + v;
}

// ---------------------------------------------------------------------------
// Pack kernels (run once per tensor; O(bytes))
// ---------------------------------------------------------------------------
// f32 [M x K] row-major -> A-fragment pack. grid = M blocks, 256 threads.
__global__ __launch_bounds__(256) void pack_a(const float* __restrict__ src, int ld,
                                              unsigned* __restrict__ dst, int K) {
  const int row = blockIdx.x, mt = row >> 4, r = row & 15, KT = K >> 5;
  const float2* s = (const float2*)(src + (size_t)row * ld);
  unsigned* d = dst + (size_t)mt * KT * 256;
  for (int kpg = threadIdx.x; kpg < (K >> 1); kpg += 256) {
    float2 v = s[kpg];
    d[(kpg >> 4) * 256 + a_slot(r, kpg & 15)] = pack_bf16x2(v.x, v.y);
  }
}

// f32 [K x N] row-major -> B-fragment pack. grid = K/2 blocks.
__global__ __launch_bounds__(256) void pack_b(const float* __restrict__ src, int ld,
                                              unsigned* __restrict__ dst, int N) {
  const int k = blockIdx.x << 1;
  const int kt = k >> 5, kp = (k >> 1) & 15, NT16 = N >> 4;
  const float* r0 = src + (size_t)k * ld;
  const float* r1 = r0 + ld;
  unsigned* d = dst + (size_t)kt * NT16 * 256;
  for (int n = threadIdx.x; n < N; n += 256)
    d[(n >> 4) * 256 + b_slot(n & 15, kp)] = pack_bf16x2(r0[n], r1[n]);
}

// f32 src [R x C] row-major -> B-fragment pack of src^T (Kdim=C, Ndim=R).
// grid = C/2 blocks.
__global__ __launch_bounds__(256) void pack_bt(const float* __restrict__ src, int ld,
                                               unsigned* __restrict__ dst, int R) {
  const int k = blockIdx.x << 1;
  const int kt = k >> 5, kp = (k >> 1) & 15, NT16 = R >> 4;
  unsigned* d = dst + (size_t)kt * NT16 * 256;
  for (int n = threadIdx.x; n < R; n += 256) {
    float2 v = *(const float2*)(src + (size_t)n * ld + k);
    d[(n >> 4) * 256 + b_slot(n & 15, kp)] = pack_bf16x2(v.x, v.y);
  }
}

// ---------------------------------------------------------------------------
// Fragment-packed GEMM:  C[M,N] (+)= Apack * Bpack (+ bias)
// Block = 128 thr (4 waves). Wave computes 32 rows x 64 cols (8 WMMAs/kt,
// 12 global_load_b128/kt, all coalesced). grid = (N/64, M/128).
// Optionally writes result as A-fragment pack (CApack) for chained GEMMs.
// ---------------------------------------------------------------------------
__global__ __launch_bounds__(128) void gemm_packed_wmma(
    const unsigned* __restrict__ Apack,
    const unsigned* __restrict__ Bpack,
    float*    __restrict__ C, int ldc,          // nullable
    const float* __restrict__ bias,             // nullable
    unsigned* __restrict__ CApack,              // nullable: bf16 A-pack of result
    int KT, int NT16, int accumulate)
{
  const int wave = threadIdx.x >> 5;
  const int lane = threadIdx.x & 31;
  const int mt0  = blockIdx.y * 8 + wave * 2;   // two 16-row fragments per wave
  const int nt0  = blockIdx.x * 4;              // four 16-col fragments

  v8f zero = {0.f,0.f,0.f,0.f,0.f,0.f,0.f,0.f};
  v8f acc[8] = {zero,zero,zero,zero,zero,zero,zero,zero};

  const unsigned* a0p = Apack + ((size_t)mt0       * KT) * 256 + lane * 8;
  const unsigned* a1p = Apack + ((size_t)(mt0 + 1) * KT) * 256 + lane * 8;
  const unsigned* bp0 = Bpack + (size_t)nt0 * 256 + lane * 8;

#pragma unroll 2
  for (int kt = 0; kt < KT; ++kt) {
    v16bf a0 = *(const v16bf*)(a0p + kt * 256);
    v16bf a1 = *(const v16bf*)(a1p + kt * 256);
    const unsigned* bp = bp0 + (size_t)kt * NT16 * 256;
    if (kt + 1 < KT) {  // pull next K-tile into WGP$/L2 (global_prefetch_b8)
      __builtin_prefetch(a0p + (kt + 1) * 256, 0, 3);
      __builtin_prefetch(bp + (size_t)NT16 * 256, 0, 3);
    }
#pragma unroll
    for (int n = 0; n < 4; ++n) {
      v16bf b = *(const v16bf*)(bp + n * 256);
      acc[n]     = __builtin_amdgcn_wmma_f32_16x16x32_bf16(false, a0, false, b, (short)0, acc[n],     false, false);
      acc[4 + n] = __builtin_amdgcn_wmma_f32_16x16x32_bf16(false, a1, false, b, (short)0, acc[4 + n], false, false);
    }
  }

  // epilogue
  const int rlo = (lane >> 4) << 3;   // 0 or 8
  const int cl  = lane & 15;
  const int KT2 = NT16 >> 1;          // K-tiles of the next gemm (N/32)
#pragma unroll
  for (int a = 0; a < 2; ++a) {
    const int mt = mt0 + a;
#pragma unroll
    for (int n = 0; n < 4; ++n) {
      const int col = (nt0 + n) * 16 + cl;
      const float bv = bias ? bias[col] : 0.0f;
#pragma unroll
      for (int r = 0; r < 8; ++r) {
        const int rin = rlo + r;              // row % 16
        float v = acc[a * 4 + n][r] + bv;
        if (C) {
          size_t off = (size_t)(mt * 16 + rin) * ldc + col;
          if (accumulate) v += C[off];
          C[off] = v;
        }
        if (CApack) {                         // wave-uniform branch
          float vodd = __shfl_xor(v, 1, 32);  // partner column value
          if ((lane & 1) == 0) {
            const int kt2 = col >> 5, kp = (col >> 1) & 15;
            CApack[((size_t)mt * KT2 + kt2) * 256 + a_slot(rin, kp)] = pack_bf16x2(v, vodd);
          }
        }
      }
    }
  }
}

// ---------------------------------------------------------------------------
// Masked softmax over 256 cols (one wave32 per row), result written directly
// as bf16 A-fragment pack (KT=8) for the following att@M / att@K GEMM.
// mask row true => all NEG_INF => uniform 1/256 (matches jax semantics).
// ---------------------------------------------------------------------------
__global__ __launch_bounds__(256) void masked_softmax_pack(
    const float* __restrict__ S, const unsigned char* __restrict__ mask,
    int L, float scale, unsigned* __restrict__ outA)
{
  const int row  = blockIdx.x * 8 + (threadIdx.x >> 5);
  const int lane = threadIdx.x & 31;
  const int b = row / L;
  const bool masked = mask[(size_t)b * L + (row - b * L)] != 0;

  const float* rowp = S + (size_t)row * 256;
  float x[8];
  float mx = -3.4e38f;
#pragma unroll
  for (int i = 0; i < 8; ++i) {
    float v = rowp[lane + i * 32] * scale;
    if (masked) v = NEG_INF_F;
    x[i] = v;
    mx = fmaxf(mx, v);
  }
#pragma unroll
  for (int off = 16; off > 0; off >>= 1) mx = fmaxf(mx, __shfl_xor(mx, off, 32));
  float s = 0.0f;
#pragma unroll
  for (int i = 0; i < 8; ++i) { x[i] = __expf(x[i] - mx); s += x[i]; }
#pragma unroll
  for (int off = 16; off > 0; off >>= 1) s += __shfl_xor(s, off, 32);
  const float inv = 1.0f / s;

  const int mt = row >> 4, r = row & 15;
  const int kp = lane >> 1;                       // lane holds k' = lane
  unsigned* dst = outA + (size_t)mt * 8 * 256 + a_slot(r, kp);
#pragma unroll
  for (int i = 0; i < 8; ++i) {
    float val = x[i] * inv;
    float vodd = __shfl_xor(val, 1, 32);
    if ((lane & 1) == 0) dst[i * 256] = pack_bf16x2(val, vodd);
  }
}

// drug_f [4096 x 512] f32 -> first half of drug_out rows (row stride 1024)
__global__ __launch_bounds__(256) void copy_drug_first_half(
    const float4* __restrict__ src, float4* __restrict__ dst)
{
  int idx = blockIdx.x * blockDim.x + threadIdx.x;   // 4096*128
  int row = idx >> 7, c = idx & 127;
  dst[(size_t)row * 256 + c] = src[idx];
}

// ---------------------------------------------------------------------------
extern "C" void kernel_launch(void* const* d_in, const int* in_sizes, int n_in,
                              void* d_out, int out_size, void* d_ws, size_t ws_size,
                              hipStream_t stream) {
  (void)in_sizes; (void)n_in; (void)out_size; (void)ws_size;

  const float* pr_f   = (const float*)d_in[0];   // [16384,1280]
  const float* drug_f = (const float*)d_in[1];   // [4096,512]
  const float* conf_p = (const float*)d_in[2];   // [256,1280]
  const float* conf_d = (const float*)d_in[3];   // [256,512]
  const float* Wq  = (const float*)d_in[4];
  const float* bq  = (const float*)d_in[5];
  const float* Wk  = (const float*)d_in[6];
  const float* bk  = (const float*)d_in[7];
  const float* Wq2 = (const float*)d_in[8];
  const float* bq2 = (const float*)d_in[9];
  const float* Wk2 = (const float*)d_in[10];
  const float* bk2 = (const float*)d_in[11];
  const float* Wpr = (const float*)d_in[12];     // [2560,512]
  const float* bpr = (const float*)d_in[13];
  const unsigned char* mask_p = (const unsigned char*)d_in[14];
  const unsigned char* mask_d = (const unsigned char*)d_in[15];

  float* pr_out = (float*)d_out;                       // [16384,512]
  float* dr_out = pr_out + (size_t)16384 * 512;        // [4096,1024]

  // ---- workspace carve (256B aligned) ----
  char* wp = (char*)d_ws;
  auto carve = [&wp](size_t bytes) -> void* {
    void* r = (void*)wp; wp += (bytes + 255) & ~(size_t)255; return r;
  };
  unsigned* prA     = (unsigned*)carve((size_t)16384 * 1280 * 2);
  unsigned* drA     = (unsigned*)carve((size_t)4096  * 512  * 2);
  unsigned* cpA     = (unsigned*)carve((size_t)256   * 1280 * 2);
  unsigned* cdA     = (unsigned*)carve((size_t)256   * 512  * 2);
  unsigned* WqB     = (unsigned*)carve((size_t)1280 * 1280 * 2);
  unsigned* WkB     = (unsigned*)carve((size_t)1280 * 1280 * 2);
  unsigned* Wq2B    = (unsigned*)carve((size_t)512  * 512  * 2);
  unsigned* Wk2B    = (unsigned*)carve((size_t)512  * 512  * 2);
  unsigned* WprTopB = (unsigned*)carve((size_t)1280 * 512  * 2);
  unsigned* WprBotB = (unsigned*)carve((size_t)1280 * 512  * 2);
  float*    Kp      = (float*)   carve((size_t)256 * 1280 * 4);
  float*    Kd      = (float*)   carve((size_t)256 * 512  * 4);
  unsigned* KpA     = (unsigned*)carve((size_t)256  * 1280 * 2);
  unsigned* KTpB    = (unsigned*)carve((size_t)1280 * 256  * 2);
  unsigned* KdB     = (unsigned*)carve((size_t)256  * 512  * 2);
  unsigned* KTdB    = (unsigned*)carve((size_t)512  * 256  * 2);
  float*    Mm      = (float*)   carve((size_t)256 * 512 * 4);
  unsigned* MmB     = (unsigned*)carve((size_t)256 * 512 * 2);
  unsigned* QpA     = (unsigned*)carve((size_t)16384 * 1280 * 2);
  unsigned* QdA     = (unsigned*)carve((size_t)4096  * 512  * 2);
  float*    Sp      = (float*)   carve((size_t)16384 * 256 * 4);
  float*    Sd      = (float*)   carve((size_t)4096  * 256 * 4);
  unsigned* SpA     = (unsigned*)carve((size_t)16384 * 256 * 2);
  unsigned* SdA     = (unsigned*)carve((size_t)4096  * 256 * 2);

  const dim3 blk128(128), blk256(256);

  // ---- one-time operand packing ----
  pack_a<<<16384, blk256, 0, stream>>>(pr_f,   1280, prA, 1280);
  pack_a<<<4096,  blk256, 0, stream>>>(drug_f, 512,  drA, 512);
  pack_a<<<256,   blk256, 0, stream>>>(conf_p, 1280, cpA, 1280);
  pack_a<<<256,   blk256, 0, stream>>>(conf_d, 512,  cdA, 512);
  pack_b<<<640, blk256, 0, stream>>>(Wq,  1280, WqB,  1280);
  pack_b<<<640, blk256, 0, stream>>>(Wk,  1280, WkB,  1280);
  pack_b<<<256, blk256, 0, stream>>>(Wq2, 512,  Wq2B, 512);
  pack_b<<<256, blk256, 0, stream>>>(Wk2, 512,  Wk2B, 512);
  pack_b<<<640, blk256, 0, stream>>>(Wpr,                    512, WprTopB, 512);
  pack_b<<<640, blk256, 0, stream>>>(Wpr + (size_t)1280*512, 512, WprBotB, 512);

  // ---- confounder keys: Kp = conf_p@Wk + bk ; Kd = conf_d@Wk2 + bk2 ----
  gemm_packed_wmma<<<dim3(20, 2), blk128, 0, stream>>>(cpA, WkB,  Kp, 1280, bk,  nullptr, 40, 80, 0);
  gemm_packed_wmma<<<dim3(8,  2), blk128, 0, stream>>>(cdA, Wk2B, Kd, 512,  bk2, nullptr, 16, 32, 0);
  pack_a <<<256, blk256, 0, stream>>>(Kp, 1280, KpA, 1280);
  pack_bt<<<640, blk256, 0, stream>>>(Kp, 1280, KTpB, 256);   // Kp^T as B
  pack_b <<<128, blk256, 0, stream>>>(Kd, 512, KdB, 512);
  pack_bt<<<256, blk256, 0, stream>>>(Kd, 512, KTdB, 256);    // Kd^T as B

  // ---- Mm = Kp @ Wpr_bot (folds Qa@Wpr_bot into att@Mm) ----
  gemm_packed_wmma<<<dim3(8, 2), blk128, 0, stream>>>(KpA, WprBotB, Mm, 512, nullptr, nullptr, 40, 32, 0);
  pack_b<<<128, blk256, 0, stream>>>(Mm, 512, MmB, 512);

  // ---- queries (bf16 fragment output only) ----
  gemm_packed_wmma<<<dim3(20, 128), blk128, 0, stream>>>(prA, WqB,  nullptr, 0, bq,  QpA, 40, 80, 0);
  gemm_packed_wmma<<<dim3(8,  32),  blk128, 0, stream>>>(drA, Wq2B, nullptr, 0, bq2, QdA, 16, 32, 0);

  // ---- attention logits + masked softmax (packed att out) ----
  gemm_packed_wmma<<<dim3(4, 128), blk128, 0, stream>>>(QpA, KTpB, Sp, 256, nullptr, nullptr, 40, 16, 0);
  masked_softmax_pack<<<2048, blk256, 0, stream>>>(Sp, mask_p, 1024, 1.0f / 16.0f, SpA);
  gemm_packed_wmma<<<dim3(4, 32), blk128, 0, stream>>>(QdA, KTdB, Sd, 256, nullptr, nullptr, 16, 16, 0);
  masked_softmax_pack<<<512, blk256, 0, stream>>>(Sd, mask_d, 256, 0.044194173824159216f, SdA);

  // ---- outputs ----
  // pr_out = pr_f @ Wpr_top + bpr ; then += att @ Mm
  gemm_packed_wmma<<<dim3(8, 128), blk128, 0, stream>>>(prA, WprTopB, pr_out, 512, bpr, nullptr, 40, 32, 0);
  gemm_packed_wmma<<<dim3(8, 128), blk128, 0, stream>>>(SpA, MmB,     pr_out, 512, nullptr, nullptr, 8, 32, 1);
  // drug_out = [drug_f | att2 @ Kd]
  copy_drug_first_half<<<2048, blk256, 0, stream>>>((const float4*)drug_f, (float4*)dr_out);
  gemm_packed_wmma<<<dim3(8, 32), blk128, 0, stream>>>(SdA, KdB, dr_out + 512, 1024, nullptr, nullptr, 8, 32, 0);
}
